// KernelAttention_54674933678709
// MI455X (gfx1250) — compile-verified, matrix operand order checked by reading, not run on hardware
//
#include <hip/hip_runtime.h>

typedef __attribute__((ext_vector_type(16))) __bf16 v16bf;
typedef __attribute__((ext_vector_type(8)))  float  v8f;
typedef __attribute__((ext_vector_type(2)))  float  v2f;

#define N_S  1024
#define N_E  512
#define N_H  8
#define N_HD 64
#define N_BS 8
#define N_C  1000
#define N_BH 64
#define LMB  0.1f
#define NB   32            // LU panel / solve block width
#define PSTR 33            // padded LDS stride for panel (bank-conflict free)
#define USTR 992           // U12 LDS stride (max trailing width)

__device__ __forceinline__ v8f wmma_bf16(v16bf a, v16bf b, v8f c) {
  // D(16x16,f32) = A(16x32,bf16) * B(32x16,bf16) + C
  return __builtin_amdgcn_wmma_f32_16x16x32_bf16(false, a, false, b, (short)0, c, false, false);
}

__device__ __forceinline__ v8f wmma_f32(v2f a, v2f b, v8f c) {
  // D(16x16,f32) = A(16x4,f32) * B(4x16,f32) + C  (full f32 for LU / solves)
  return __builtin_amdgcn_wmma_f32_16x16x4_f32(false, a, false, b, (short)0, c, false, false);
}

// Per-lane K-coordinate for 16-bit 16x32 A fragment (ISA 7.12.2)
__device__ __forceinline__ int frag_k(int lane, int e) {
  int half = lane >> 4;
  int d = e >> 1, s = e & 1;
  return (d < 4) ? (2 * d + s + 8 * half) : (16 + 2 * (d - 4) + s + 8 * half);
}

// ---------------- Stage 1: q = x * Wi^T + bi, stored head-split [(b*H+h)*S+s]*HD+d
__global__ void proj_q(const float* __restrict__ x, const float* __restrict__ Wi,
                       const float* __restrict__ bi, float* __restrict__ q) {
  int mt = blockIdx.x, nt = blockIdx.y;
  int lane = threadIdx.x, row = lane & 15, half = lane >> 4;
  v8f acc = {};
  for (int k0 = 0; k0 < N_E; k0 += 32) {
    v16bf a, b;
#pragma unroll
    for (int e = 0; e < 16; ++e) {
      int kk = frag_k(lane, e);
      a[e] = (__bf16)x[(size_t)(mt * 16 + row) * N_E + k0 + kk];
      b[e] = (__bf16)Wi[(size_t)(nt * 16 + row) * N_E + k0 + kk]; // B[k][n] = Wi[n][k]
    }
    acc = wmma_bf16(a, b, acc);
  }
#pragma unroll
  for (int r = 0; r < 8; ++r) {
    int m = mt * 16 + r + 8 * half;
    int n = nt * 16 + row;
    int bb = m >> 10, s = m & 1023;
    int h = n >> 6, d = n & 63;
    q[(((size_t)bb * N_H + h) * N_S + s) * N_HD + d] = acc[r] + bi[n];
  }
}

// ---------------- Stage 2: sq[bh][s] = ||q/ls||^2
__global__ void row_sq(const float* __restrict__ q, const float* __restrict__ ls,
                       float* __restrict__ sq) {
  int idx = blockIdx.x * blockDim.x + threadIdx.x;
  float inv_ls = 1.0f / ls[0];
  const float* qr = q + (size_t)idx * N_HD;
  float acc = 0.f;
#pragma unroll
  for (int d = 0; d < N_HD; ++d) { float v = qr[d]; acc += v * v; }
  sq[idx] = acc * inv_ls * inv_ls;
}

// ---------------- Stage 3: A = exp(-0.5*max(d2,0)) - lambda*I (Gram via WMMA)
__global__ void build_A(const float* __restrict__ q, const float* __restrict__ sq,
                        const float* __restrict__ ls, float* __restrict__ A) {
  int bh = blockIdx.y;
  int it = blockIdx.x >> 6, jt = blockIdx.x & 63;
  int lane = threadIdx.x, row = lane & 15, half = lane >> 4;
  float inv_ls = 1.0f / ls[0];
  const float* qb  = q  + (size_t)bh * N_S * N_HD;
  const float* sqb = sq + (size_t)bh * N_S;
  v8f g = {};
  for (int k0 = 0; k0 < N_HD; k0 += 32) {
    v16bf a, b;
#pragma unroll
    for (int e = 0; e < 16; ++e) {
      int kk = frag_k(lane, e);
      a[e] = (__bf16)(qb[(size_t)(it * 16 + row) * N_HD + k0 + kk] * inv_ls);
      b[e] = (__bf16)(qb[(size_t)(jt * 16 + row) * N_HD + k0 + kk] * inv_ls);
    }
    g = wmma_bf16(a, b, g);
  }
  float* Ab = A + (size_t)bh * N_S * N_S;
#pragma unroll
  for (int r = 0; r < 8; ++r) {
    int i = it * 16 + r + 8 * half;
    int j = jt * 16 + row;
    float d2 = sqb[i] + sqb[j] - 2.0f * g[r];
    float kv = __expf(-0.5f * fmaxf(d2, 0.0f));
    Ab[(size_t)i * N_S + j] = kv - (i == j ? LMB : 0.0f);
  }
}

// ---------------- Stage 4: blocked in-place LU (NB=32), one workgroup per matrix.
// Panel + U12 live in LDS (256KB of the 320KB WGP pool); trailing update is a
// K=32 GEMM on V_WMMA_F32_16X16X4_F32 with both operands in LDS.
__global__ __launch_bounds__(1024) void lu_fact(float* __restrict__ A) {
  extern __shared__ float lds[];
  float* pan = lds;                // [N_S][PSTR] : panel / L21 (135KB)
  float* u12 = lds + N_S * PSTR;   // [NB][USTR]  : U12 row block (124KB)
  int bh = blockIdx.x, tid = threadIdx.x;
  int wave = tid >> 5, lane = tid & 31, row = lane & 15, half = lane >> 4;
  float* Ab = A + (size_t)bh * N_S * N_S;

  for (int k0 = 0; k0 < N_S; k0 += NB) {
    int R = N_S - k0;              // panel height
    int W = R - NB;                // trailing width (multiple of 32, may be 0)

    // ---- load panel A[k0:,k0:k0+NB] into LDS (coalesced) ----
    for (int idx = tid; idx < R * NB; idx += 1024) {
      int r = idx >> 5, c = idx & 31;
      pan[r * PSTR + c] = Ab[(size_t)(k0 + r) * N_S + k0 + c];
    }
    __syncthreads();

    // ---- unblocked factorization of the panel, entirely in LDS ----
    for (int c = 0; c < NB; ++c) {
      float ip = 1.0f / pan[c * PSTR + c];
      for (int r = c + 1 + tid; r < R; r += 1024) {
        float lik = pan[r * PSTR + c] * ip;
        pan[r * PSTR + c] = lik;
        for (int cc = c + 1; cc < NB; ++cc)
          pan[r * PSTR + cc] -= lik * pan[c * PSTR + cc];
      }
      __syncthreads();
    }

    // ---- write panel back ----
    for (int idx = tid; idx < R * NB; idx += 1024) {
      int r = idx >> 5, c = idx & 31;
      Ab[(size_t)(k0 + r) * N_S + k0 + c] = pan[r * PSTR + c];
    }

    if (W > 0) {
      // ---- U12 = L11^{-1} * A12 : one column per thread, in registers ----
      if (tid < W) {
        int j = k0 + NB + tid;
        float v[NB];
#pragma unroll
        for (int i = 0; i < NB; ++i) v[i] = Ab[(size_t)(k0 + i) * N_S + j];
#pragma unroll
        for (int i = 1; i < NB; ++i) {
          float acc = v[i];
#pragma unroll
          for (int m = 0; m < i; ++m) acc -= pan[i * PSTR + m] * v[m];
          v[i] = acc;
        }
#pragma unroll
        for (int i = 0; i < NB; ++i) {
          u12[i * USTR + tid] = v[i];
          Ab[(size_t)(k0 + i) * N_S + j] = v[i];
        }
      }
      __syncthreads();

      // ---- trailing update A22 -= L21 * U12 via f32 WMMA (16x16 tiles) ----
      int nt = W >> 4;
      for (int t = wave; t < nt * nt; t += 32) {   // wave-uniform loop: EXEC all-1s
        int ti = t / nt, tj = t % nt;
        int i0 = k0 + NB + ti * 16, j0 = k0 + NB + tj * 16;
        v8f cf;
#pragma unroll
        for (int r = 0; r < 8; ++r)
          cf[r] = Ab[(size_t)(i0 + r + 8 * half) * N_S + j0 + row];
#pragma unroll
        for (int kc = 0; kc < NB; kc += 4) {
          v2f af, bf;
#pragma unroll
          for (int d = 0; d < 2; ++d) {
            int kk = kc + d + 2 * half;                    // 16x4 f32 A layout
            af[d] = -pan[(NB + ti * 16 + row) * PSTR + kk]; // fold the subtraction
            bf[d] =  u12[kk * USTR + tj * 16 + row];
          }
          cf = wmma_f32(af, bf, cf);
        }
#pragma unroll
        for (int r = 0; r < 8; ++r)
          Ab[(size_t)(i0 + r + 8 * half) * N_S + j0 + row] = cf[r];
      }
    }
    __syncthreads();
  }
}

// ---------------- Stage 5: blocked triangular solves (LU) y = q.
// 1024x64 RHS resident in LDS (256KB). Per 32-row block the bulk update
// T = blk - L[blk,0:i0] @ z  (resp. U side) is a 32x64xK GEMM on f32 WMMA:
// 8 waves <-> the 8 16x16 output tiles, wave-uniform K loop. Only the 32x32
// in-block substitution (496 FMAs) stays serial.
__global__ __launch_bounds__(256) void tri_solve(const float* __restrict__ A,
                                                 const float* __restrict__ q,
                                                 float* __restrict__ y) {
  extern __shared__ float z[]; // N_S * N_HD floats; z[j*64+c] -> bank c
  int bh = blockIdx.x, tid = threadIdx.x;
  int wave = tid >> 5, lane = tid & 31, row = lane & 15, half = lane >> 4;
  int rt = wave >> 2, ct = wave & 3;     // 2x4 tiles of the 32x64 block
  const float* Ab = A + (size_t)bh * N_S * N_S;
  const float* qb = q + (size_t)bh * N_S * N_HD;

  // ---- forward: L z = q (unit lower) ----
  for (int bi = 0; bi < N_S / NB; ++bi) {
    int i0 = bi * NB;
    v8f acc;
#pragma unroll
    for (int r = 0; r < 8; ++r)
      acc[r] = qb[(size_t)(i0 + rt * 16 + r + 8 * half) * N_HD + ct * 16 + row];
    for (int k = 0; k < i0; k += 4) {     // K = 32*bi, wave-uniform
      v2f af, bf;
#pragma unroll
      for (int d = 0; d < 2; ++d) {
        int kk = k + d + 2 * half;
        af[d] = -Ab[(size_t)(i0 + rt * 16 + row) * N_S + kk];
        bf[d] = z[kk * N_HD + ct * 16 + row];
      }
      acc = wmma_f32(af, bf, acc);
    }
#pragma unroll
    for (int r = 0; r < 8; ++r)           // each wave writes only its own tile
      z[(i0 + rt * 16 + r + 8 * half) * N_HD + ct * 16 + row] = acc[r];
    __syncthreads();
    if (tid < N_HD) {                     // in-block substitution, one column/thread
      int c = tid;
      for (int i = 1; i < NB; ++i) {
        float av = z[(i0 + i) * N_HD + c];
        for (int m = 0; m < i; ++m)
          av -= Ab[(size_t)(i0 + i) * N_S + i0 + m] * z[(i0 + m) * N_HD + c];
        z[(i0 + i) * N_HD + c] = av;
      }
    }
    __syncthreads();
  }

  // ---- backward: U y = z (non-unit diag) ----
  for (int bi = N_S / NB - 1; bi >= 0; --bi) {
    int i0 = bi * NB;
    v8f acc;
#pragma unroll
    for (int r = 0; r < 8; ++r)
      acc[r] = z[(i0 + rt * 16 + r + 8 * half) * N_HD + ct * 16 + row];
    for (int k = i0 + NB; k < N_S; k += 4) {   // K = 32*(31-bi), wave-uniform
      v2f af, bf;
#pragma unroll
      for (int d = 0; d < 2; ++d) {
        int kk = k + d + 2 * half;
        af[d] = -Ab[(size_t)(i0 + rt * 16 + row) * N_S + kk];
        bf[d] = z[kk * N_HD + ct * 16 + row];
      }
      acc = wmma_f32(af, bf, acc);
    }
#pragma unroll
    for (int r = 0; r < 8; ++r)
      z[(i0 + rt * 16 + r + 8 * half) * N_HD + ct * 16 + row] = acc[r];
    __syncthreads();
    if (tid < N_HD) {
      int c = tid;
      for (int i = NB - 1; i >= 0; --i) {
        float av = z[(i0 + i) * N_HD + c];
        for (int m = i + 1; m < NB; ++m)
          av -= Ab[(size_t)(i0 + i) * N_S + i0 + m] * z[(i0 + m) * N_HD + c];
        float yv = av / Ab[(size_t)(i0 + i) * N_S + i0 + i];
        z[(i0 + i) * N_HD + c] = yv;
        y[((size_t)bh * N_S + i0 + i) * N_HD + c] = yv;
      }
    }
    __syncthreads();
  }
}

// ---------------- Stage 6: o1 = K @ y with K tiles recomputed on the fly
__global__ void kxy(const float* __restrict__ q, const float* __restrict__ sq,
                    const float* __restrict__ ls, const float* __restrict__ y,
                    float* __restrict__ o1) {
  __shared__ float ktile[16 * 32];
  int it = blockIdx.x, bh = blockIdx.y;
  int lane = threadIdx.x, row = lane & 15, half = lane >> 4;
  float inv_ls = 1.0f / ls[0];
  const float* qb  = q  + (size_t)bh * N_S * N_HD;
  const float* sqb = sq + (size_t)bh * N_S;
  const float* yb  = y  + (size_t)bh * N_S * N_HD;
  v8f acc[4] = {};
  float sqi[8];
#pragma unroll
  for (int r = 0; r < 8; ++r) sqi[r] = sqb[it * 16 + r + 8 * half];
  for (int j0 = 0; j0 < N_S; j0 += 32) {
    v8f g0 = {}, g1 = {};
    for (int k0 = 0; k0 < N_HD; k0 += 32) {
      v16bf a, b0, b1;
#pragma unroll
      for (int e = 0; e < 16; ++e) {
        int kk = frag_k(lane, e);
        a[e]  = (__bf16)(qb[(size_t)(it * 16 + row) * N_HD + k0 + kk] * inv_ls);
        b0[e] = (__bf16)(qb[(size_t)(j0 + row) * N_HD + k0 + kk] * inv_ls);
        b1[e] = (__bf16)(qb[(size_t)(j0 + 16 + row) * N_HD + k0 + kk] * inv_ls);
      }
      g0 = wmma_bf16(a, b0, g0);
      g1 = wmma_bf16(a, b1, g1);
    }
    float sj0 = sqb[j0 + row], sj1 = sqb[j0 + 16 + row];
#pragma unroll
    for (int r = 0; r < 8; ++r) {
      float k0v = __expf(-0.5f * fmaxf(sqi[r] + sj0 - 2.0f * g0[r], 0.0f));
      float k1v = __expf(-0.5f * fmaxf(sqi[r] + sj1 - 2.0f * g1[r], 0.0f));
      ktile[(r + 8 * half) * 32 + row]      = k0v;
      ktile[(r + 8 * half) * 32 + 16 + row] = k1v;
    }
    __syncthreads();
    v16bf ka;
#pragma unroll
    for (int e = 0; e < 16; ++e) ka[e] = (__bf16)ktile[row * 32 + frag_k(lane, e)];
#pragma unroll
    for (int nt = 0; nt < 4; ++nt) {
      v16bf yf;
#pragma unroll
      for (int e = 0; e < 16; ++e) {
        int kk = frag_k(lane, e);
        yf[e] = (__bf16)yb[(size_t)(j0 + kk) * N_HD + nt * 16 + row];
      }
      acc[nt] = wmma_bf16(ka, yf, acc[nt]);
    }
    __syncthreads();
  }
#pragma unroll
  for (int nt = 0; nt < 4; ++nt)
#pragma unroll
    for (int r = 0; r < 8; ++r)
      o1[((size_t)bh * N_S + it * 16 + r + 8 * half) * N_HD + nt * 16 + row] = acc[nt][r];
}

// ---------------- Stage 7: out = o1_flat * Wo^T + bo
__global__ void proj_out(const float* __restrict__ o1, const float* __restrict__ Wo,
                         const float* __restrict__ bo, float* __restrict__ out) {
  int mt = blockIdx.x, nt = blockIdx.y;
  int lane = threadIdx.x, row = lane & 15, half = lane >> 4;
  v8f acc = {};
  for (int k0 = 0; k0 < N_E; k0 += 32) {
    v16bf a, b;
#pragma unroll
    for (int e = 0; e < 16; ++e) {
      int kk = frag_k(lane, e);
      a[e] = (__bf16)o1[(size_t)(mt * 16 + row) * N_E + k0 + kk];
      int n = nt * 16 + row;
      b[e] = (n < N_C) ? (__bf16)Wo[(size_t)n * N_E + k0 + kk] : (__bf16)0.0f;
    }
    acc = wmma_bf16(a, b, acc);
  }
#pragma unroll
  for (int r = 0; r < 8; ++r) {
    int m = mt * 16 + r + 8 * half;
    int n = nt * 16 + row;
    if (n < N_C) out[(size_t)m * N_C + n] = acc[r] + bo[n];
  }
}

extern "C" void kernel_launch(void* const* d_in, const int* in_sizes, int n_in,
                              void* d_out, int out_size, void* d_ws, size_t ws_size,
                              hipStream_t stream) {
  const float* x  = (const float*)d_in[0];
  const float* Wi = (const float*)d_in[1];
  const float* bi = (const float*)d_in[2];
  const float* Wo = (const float*)d_in[3];
  const float* bo = (const float*)d_in[4];
  const float* ls = (const float*)d_in[5];
  float* out = (float*)d_out;

  float* ws = (float*)d_ws;
  float* q  = ws;              // 4,194,304 f32 (16 MB)
  float* y  = ws + 4194304;    // 4,194,304
  float* o1 = ws + 8388608;    // 4,194,304
  float* A  = ws + 12582912;   // 67,108,864 (256 MB)
  float* sq = ws + 79691776;   // 65,536

  size_t lu_lds = (size_t)(N_S * PSTR + NB * USTR) * sizeof(float); // 262,144 B
  size_t ts_lds = (size_t)N_S * N_HD * sizeof(float);               // 262,144 B

  proj_q   <<<dim3(N_BS * N_S / 16, N_E / 16), 32, 0, stream>>>(x, Wi, bi, q);
  row_sq   <<<N_BH * N_S / 256, 256, 0, stream>>>(q, ls, sq);
  build_A  <<<dim3((N_S / 16) * (N_S / 16), N_BH), 32, 0, stream>>>(q, sq, ls, A);
  lu_fact  <<<N_BH, 1024, lu_lds, stream>>>(A);
  tri_solve<<<N_BH, 256, ts_lds, stream>>>(A, q, y);
  kxy      <<<dim3(N_S / 16, N_BH), 32, 0, stream>>>(q, sq, ls, y, o1);
  proj_out <<<dim3(N_BS * N_S / 16, (N_C + 15) / 16), 32, 0, stream>>>(o1, Wo, bo, out);
}
